// RetNetDecoderLayer_70239895158923
// MI455X (gfx1250) — matrix operand order repaired and checked by reading, not compiled, not run on hardware
//
#include <hip/hip_runtime.h>
#include <hip/hip_bf16.h>

typedef unsigned short ushort_t;
typedef unsigned int uint_t;
typedef unsigned long long u64_t;

typedef __attribute__((ext_vector_type(16))) __bf16 bf16x16;
typedef __attribute__((ext_vector_type(8)))  float  f32x8;
typedef __attribute__((ext_vector_type(2)))  float  f32x2;
typedef __attribute__((ext_vector_type(4)))  uint_t u32x4;
typedef __attribute__((ext_vector_type(8)))  int    i32x8;
typedef __attribute__((ext_vector_type(4)))  int    i32x4;

// ---------------- problem constants ----------------
constexpr int B_  = 2;
constexpr int T_  = 2048;
constexpr int D_  = 2048;
constexpr int H_  = 8;
constexpr int DK_ = 256;
constexpr int DV_ = 256;
constexpr int N_  = B_ * T_;       // 4096 rows
constexpr int INTER_ = 2816;
constexpr int CHUNK_ = 64;
constexpr int NCHUNK_ = T_ / CHUNK_;
constexpr float EPS_ = 1e-5f;

__device__ __forceinline__ ushort_t f2bf(float f) {
    uint_t u = __float_as_uint(f);
    u += 0x7FFFu + ((u >> 16) & 1u);     // round-to-nearest-even
    return (ushort_t)(u >> 16);
}
__device__ __forceinline__ float siluf(float x) { return x / (1.f + expf(-x)); }
__device__ __forceinline__ float sigmf(float x) { return 1.f / (1.f + expf(-x)); }

union BF16Frag { uint_t u[8]; bf16x16 v; };

// =====================================================================
// TDM: issue 2-D tile load global->LDS (Tensor Data Mover, TENSORcnt)
//   tile is tile_rows x 32 bf16 elements, row-major in global,
//   landing in LDS with pitch 40 ushorts (pad 4 dwords every 16 dwords).
//   (6-arg builtin variant: g0, g1, g2, g3, g4(zero), cpol)
// =====================================================================
__device__ __forceinline__ void tdm_load_tileK32(uint_t lds_off, const ushort_t* gtile,
                                                 int tensor_d0, int tensor_d1,
                                                 int tile_rows) {
    u32x4 g0;
    u64_t ga = (u64_t)(uintptr_t)gtile;
    g0.x = 1u;                                    // count=1, user mode, no gather
    g0.y = lds_off;                               // LDS byte address
    g0.z = (uint_t)(ga & 0xFFFFFFFFu);            // global addr lo
    g0.w = (uint_t)(ga >> 32) | (2u << 30);       // global addr hi | type=2 (image)
    i32x8 g1;
    // data_size=1 (2 bytes) | pad_enable | pad_interval=3 (16 dw) | pad_amount=3 (4 dw)
    g1[0] = (1 << 16) | (1 << 20) | (3 << 22) | (3 << 25);
    g1[1] = (int)(((uint_t)tensor_d0 & 0xFFFFu) << 16);                  // dim0 lo16
    g1[2] = (int)(((uint_t)tensor_d0 >> 16) | (((uint_t)tensor_d1 & 0xFFFFu) << 16));
    g1[3] = (int)(((uint_t)tensor_d1 >> 16) | (32u << 16));              // dim1 hi | tile_dim0=32
    g1[4] = (int)((uint_t)tile_rows & 0xFFFFu);                          // tile_dim1 ; tile_dim2=0
    g1[5] = tensor_d0;                                                   // dim0 stride lo32
    g1[6] = 0;
    g1[7] = 0;
    i32x4 z4 = {0, 0, 0, 0};
    i32x8 z8 = {0, 0, 0, 0, 0, 0, 0, 0};
    __builtin_amdgcn_tensor_load_to_lds(g0, g1, z4, z4, z8, 0);
}

// =====================================================================
// 1) rmsnorm (fp32 in) -> bf16 out, one block per row of D_ elems
// =====================================================================
__global__ __launch_bounds__(256) void rmsnorm_to_bf16_k(const float* __restrict__ x,
                                                         const float* __restrict__ w,
                                                         ushort_t* __restrict__ out) {
    __shared__ float red[256];
    const int row = blockIdx.x;
    const int tid = threadIdx.x;
    const float* xr = x + (size_t)row * D_;
    float s = 0.f;
    float vals[D_ / 256];
#pragma unroll
    for (int e = 0; e < D_ / 256; ++e) {
        float v = xr[tid + e * 256];
        vals[e] = v;
        s += v * v;
    }
    red[tid] = s; __syncthreads();
    for (int st = 128; st > 0; st >>= 1) {
        if (tid < st) red[tid] += red[tid + st];
        __syncthreads();
    }
    const float rs = rsqrtf(red[0] / (float)D_ + EPS_);
    ushort_t* orow = out + (size_t)row * D_;
#pragma unroll
    for (int e = 0; e < D_ / 256; ++e) {
        int c = tid + e * 256;
        orow[c] = f2bf(vals[e] * rs * w[c]);
    }
}

// =====================================================================
// 2) fp32 -> bf16 transposing convert (weights -> WT[n][k])
//    grid (C/32, R/32), 256 threads
// =====================================================================
__global__ __launch_bounds__(256) void transpose_to_bf16_k(const float* __restrict__ in,
                                                           ushort_t* __restrict__ out,
                                                           int R, int C) {
    __shared__ ushort_t t[32][33];
    const int c0 = blockIdx.x * 32, r0 = blockIdx.y * 32;
    const int tx = threadIdx.x & 31, ty = threadIdx.x >> 5;
#pragma unroll
    for (int e = 0; e < 4; ++e) {
        int r = r0 + ty + e * 8;
        t[ty + e * 8][tx] = f2bf(in[(size_t)r * C + c0 + tx]);
    }
    __syncthreads();
#pragma unroll
    for (int e = 0; e < 4; ++e) {
        int c = c0 + ty + e * 8;
        out[(size_t)c * R + r0 + tx] = t[tx][ty + e * 8];
    }
}

// =====================================================================
// 3) bf16 GEMM: C[M,N] = A[M,K] @ W[K,N] with W given TRANSPOSED (Wt[N,K]).
//    128x128 block tile, BK=32, 8 waves, wave tile 32x64 (2x4 WMMA tiles).
//    Tiles staged by the Tensor Data Mover (wave 0), double buffered.
// =====================================================================
__global__ __launch_bounds__(256) void gemm_bf16_k(const ushort_t* __restrict__ A,
                                                   const ushort_t* __restrict__ Wt,
                                                   float* __restrict__ C,
                                                   int M, int N, int K) {
    __shared__ ushort_t As[2][128][40];   // [m][k], pitch 40 (TDM-padded)
    __shared__ ushort_t Bs[2][128][40];   // [n][k], pitch 40 (TDM-padded)

    const int tid  = threadIdx.x;
    const int lane = tid & 31;
    const int wv   = tid >> 5;
    const int mw   = (wv & 3) * 32;
    const int nw   = (wv >> 2) * 64;
    const int rowBlk = blockIdx.y * 128;
    const int colBlk = blockIdx.x * 128;
    const int nlo  = lane & 15;
    const bool hi  = lane >= 16;

    const ushort_t* Abase = A  + (size_t)rowBlk * K;
    const ushort_t* Bbase = Wt + (size_t)colBlk * K;

    f32x8 acc[2][4] = {};

    const int nIter = K / 32;
    if (wv == 0) {
        tdm_load_tileK32((uint_t)(uintptr_t)&As[0][0][0], Abase, K, M, 128);
        tdm_load_tileK32((uint_t)(uintptr_t)&Bs[0][0][0], Bbase, K, N, 128);
    }

    for (int it = 0; it < nIter; ++it) {
        const int cur = it & 1;
        if (wv == 0) {
            if (it + 1 < nIter) {
                const int nxt = cur ^ 1;
                tdm_load_tileK32((uint_t)(uintptr_t)&As[nxt][0][0],
                                 Abase + (it + 1) * 32, K, M, 128);
                tdm_load_tileK32((uint_t)(uintptr_t)&Bs[nxt][0][0],
                                 Bbase + (it + 1) * 32, K, N, 128);
                __builtin_amdgcn_s_wait_tensorcnt((short)2);
            } else {
                __builtin_amdgcn_s_wait_tensorcnt((short)0);
            }
        }
        __syncthreads();

        BF16Frag a[2], b[4];
#pragma unroll
        for (int ti = 0; ti < 2; ++ti) {
            int m = mw + ti * 16 + nlo;
#pragma unroll
            for (int j = 0; j < 8; ++j) {
                int kb = ((j >> 2) << 4) + (hi ? 8 : 0) + ((j & 3) << 1);
                a[ti].u[j] = *(const uint_t*)&As[cur][m][kb];
            }
        }
#pragma unroll
        for (int tj = 0; tj < 4; ++tj) {
            int col = nw + tj * 16 + nlo;
#pragma unroll
            for (int j = 0; j < 8; ++j) {
                int kb = ((j >> 2) << 4) + (hi ? 8 : 0) + ((j & 3) << 1);
                b[tj].u[j] = *(const uint_t*)&Bs[cur][col][kb];
            }
        }
#pragma unroll
        for (int ti = 0; ti < 2; ++ti)
#pragma unroll
            for (int tj = 0; tj < 4; ++tj)
                acc[ti][tj] = __builtin_amdgcn_wmma_f32_16x16x32_bf16(
                    false, a[ti].v, false, b[tj].v, (short)0, acc[ti][tj], false, false);
        __syncthreads();
    }

#pragma unroll
    for (int ti = 0; ti < 2; ++ti)
#pragma unroll
        for (int tj = 0; tj < 4; ++tj)
#pragma unroll
            for (int r = 0; r < 8; ++r) {
                int row = rowBlk + mw + ti * 16 + (hi ? r + 8 : r);
                int col = colBlk + nw + tj * 16 + nlo;
                C[(size_t)row * N + col] = acc[ti][tj][r];
            }
}

// =====================================================================
// 4) short conv (K=4, causal) + silu, elementwise over [B,T,D]
// =====================================================================
__global__ __launch_bounds__(256) void conv_silu_k(const float* __restrict__ pre,
                                                   const float* __restrict__ cw,
                                                   float* __restrict__ out) {
    size_t idx = (size_t)blockIdx.x * blockDim.x + threadIdx.x;
    if (idx >= (size_t)N_ * D_) return;
    int d = (int)(idx % D_);
    int t = (int)((idx / D_) % T_);
    int b = (int)(idx / ((size_t)T_ * D_));
    float y = 0.f;
#pragma unroll
    for (int i = 0; i < 4; ++i) {
        int tt = t - 3 + i;
        if (tt >= 0) y += pre[((size_t)(b * T_ + tt)) * D_ + d] * cw[d * 4 + i];
    }
    out[idx] = siluf(y);
}

// =====================================================================
// 5) rope in place on [B,T,H,DK] (d = h*DK + i)
// =====================================================================
__global__ __launch_bounds__(256) void rope_k(float* __restrict__ x) {
    size_t idx = (size_t)blockIdx.x * blockDim.x + threadIdx.x;   // B*T*H*128
    if (idx >= (size_t)B_ * T_ * H_ * (DK_ / 2)) return;
    int i = (int)(idx % (DK_ / 2));
    int h = (int)((idx / (DK_ / 2)) % H_);
    int t = (int)((idx / ((DK_ / 2) * H_)) % T_);
    int b = (int)(idx / ((size_t)(DK_ / 2) * H_ * T_));
    float inv = exp2f(-(float)i * (13.2877123795494f / 128.f));   // 10000^(-i/128)
    float fr = (float)t * inv;
    float c = cosf(fr), s = sinf(fr);
    size_t base = ((size_t)(b * T_ + t)) * D_ + h * DK_ + i;
    float x1 = x[base], x2 = x[base + 128];
    x[base]       = x1 * c - x2 * s;
    x[base + 128] = x2 * c + x1 * s;
}

// =====================================================================
// 6) chunked retention with fp32 WMMA (16x16x4)
//    grid = B*H*(DV/32) = 128 blocks of 256 threads (8 waves)
// =====================================================================
__global__ __launch_bounds__(256) void retention_k(const float* __restrict__ q,
                                                   const float* __restrict__ k,
                                                   const float* __restrict__ v,
                                                   float* __restrict__ o) {
    const int blk = blockIdx.x;
    const int vs = blk & 7;            // V slice (32 wide)
    const int h  = (blk >> 3) & 7;
    const int b  = blk >> 6;
    const float gamma = 1.f - exp2f(-5.f - (float)h);
    const float lg = log2f(gamma);
    const float gC = exp2f(64.f * lg);

    const int lane = threadIdx.x & 31;
    const int wv   = threadIdx.x >> 5;
    const int nlo  = lane & 15;
    const bool hi  = lane >= 16;
    const int koff = hi ? 2 : 0;
    const int mrel = hi ? 8 : 0;

    __shared__ float S[DK_][36];       // state slice [dk][dv32], padded
    __shared__ float Am[64][68];       // masked qk^T, padded
    __shared__ float gp[64];           // gamma^i table

    for (int i = threadIdx.x; i < DK_ * 36; i += 256) (&S[0][0])[i] = 0.f;
    if (threadIdx.x < 64) gp[threadIdx.x] = exp2f((float)threadIdx.x * lg);
    __syncthreads();

    const size_t qkbase = ((size_t)b * T_) * D_ + h * DK_;
    const int vcol = vs * 32;

    for (int c = 0; c < NCHUNK_; ++c) {
        const int t0 = c * CHUNK_;

        // ---- Phase 1: Am = (q * dk^-0.5) @ k^T with decay mask. 2 tiles/wave
#pragma unroll
        for (int s = 0; s < 2; ++s) {
            int idx = wv * 2 + s;
            int mi = idx >> 2, ni = idx & 3;
            f32x8 acc = {};
            const float* qp = q + qkbase + (size_t)(t0 + mi * 16 + nlo) * D_;
            const float* kp = k + qkbase + (size_t)(t0 + ni * 16 + nlo) * D_;
            for (int kk = 0; kk < DK_; kk += 4) {
                f32x2 a, bb;
                a.x = qp[kk + koff]     * 0.0625f;
                a.y = qp[kk + koff + 1] * 0.0625f;
                bb.x = kp[kk + koff];
                bb.y = kp[kk + koff + 1];
                acc = __builtin_amdgcn_wmma_f32_16x16x4_f32(false, a, false, bb,
                                                            (short)0, acc, false, false);
            }
#pragma unroll
            for (int r = 0; r < 8; ++r) {
                int m = mi * 16 + mrel + r;
                int n = ni * 16 + nlo;
                int diff = m - n;
                Am[m][n] = (diff >= 0) ? acc[r] * gp[diff] : 0.f;
            }
        }
        __syncthreads();

        // ---- Phase 2: o = Am @ v_slice + (q @ S) * di. 1 tile/wave
        {
            int mi = wv & 3, ni = wv >> 2;
            f32x8 accS = {};
            const float* qp = q + qkbase + (size_t)(t0 + mi * 16 + nlo) * D_;
            for (int kk = 0; kk < DK_; kk += 4) {
                f32x2 a, bb;
                a.x = qp[kk + koff]     * 0.0625f;
                a.y = qp[kk + koff + 1] * 0.0625f;
                bb.x = S[kk + koff][ni * 16 + nlo];
                bb.y = S[kk + koff + 1][ni * 16 + nlo];
                accS = __builtin_amdgcn_wmma_f32_16x16x4_f32(false, a, false, bb,
                                                             (short)0, accS, false, false);
            }
            f32x8 accA = {};
            const float* Arow = &Am[mi * 16 + nlo][0];
            for (int kk = 0; kk < CHUNK_; kk += 4) {
                int j0 = kk + koff;
                f32x2 a, bb;
                a.x = Arow[j0];
                a.y = Arow[j0 + 1];
                const float* vp = v + qkbase + (size_t)(t0 + j0) * D_ + vcol + ni * 16 + nlo;
                bb.x = vp[0];
                bb.y = vp[D_];
                accA = __builtin_amdgcn_wmma_f32_16x16x4_f32(false, a, false, bb,
                                                             (short)0, accA, false, false);
            }
#pragma unroll
            for (int r = 0; r < 8; ++r) {
                int m = mi * 16 + mrel + r;
                float di = gp[m] * gamma;   // gamma^(m+1)
                o[((size_t)(b * T_) + t0 + m) * D_ + h * DV_ + vcol + ni * 16 + nlo]
                    = accA[r] + accS[r] * di;
            }
        }
        __syncthreads();

        // ---- Phase 3: S = S*gamma^C + (k*dkc)^T @ v_slice. 4 tiles/wave
        f32x8 upd[4];
#pragma unroll
        for (int s = 0; s < 4; ++s) {
            int idx = wv * 4 + s;
            int ti = idx >> 1, tj = idx & 1;
            f32x8 acc;
#pragma unroll
            for (int r = 0; r < 8; ++r)
                acc[r] = S[ti * 16 + mrel + r][tj * 16 + nlo] * gC;
            int dk = ti * 16 + nlo;
            for (int kk = 0; kk < CHUNK_; kk += 4) {
                int j0 = kk + koff;
                const float* kp = k + qkbase + (size_t)(t0 + j0) * D_ + dk;
                const float* vp = v + qkbase + (size_t)(t0 + j0) * D_ + vcol + tj * 16 + nlo;
                f32x2 a, bb;
                a.x = kp[0]  * gp[63 - j0];
                a.y = kp[D_] * gp[62 - j0];
                bb.x = vp[0];
                bb.y = vp[D_];
                acc = __builtin_amdgcn_wmma_f32_16x16x4_f32(false, a, false, bb,
                                                            (short)0, acc, false, false);
            }
            upd[s] = acc;
        }
#pragma unroll
        for (int s = 0; s < 4; ++s) {
            int idx = wv * 4 + s;
            int ti = idx >> 1, tj = idx & 1;
#pragma unroll
            for (int r = 0; r < 8; ++r)
                S[ti * 16 + mrel + r][tj * 16 + nlo] = upd[s][r];
        }
    }
}

// =====================================================================
// 7) per-head group rmsnorm of o + silu(g) gate -> bf16
// =====================================================================
__global__ __launch_bounds__(256) void gated_out_k(const float* __restrict__ o,
                                                   const float* __restrict__ g,
                                                   const float* __restrict__ gw,
                                                   ushort_t* __restrict__ out) {
    const int n = blockIdx.x;
    const int lane = threadIdx.x & 31;
    const int h = threadIdx.x >> 5;
    const float* orow = o + (size_t)n * D_ + h * DV_;
    const float* grow = g + (size_t)n * D_ + h * DV_;
    float vals[8];
    float s = 0.f;
#pragma unroll
    for (int e = 0; e < 8; ++e) {
        float v = orow[lane + e * 32];
        vals[e] = v;
        s += v * v;
    }
#pragma unroll
    for (int m = 16; m >= 1; m >>= 1) s += __shfl_xor(s, m, 32);
    const float rs = rsqrtf(s / (float)DV_ + EPS_);
    ushort_t* outr = out + (size_t)n * D_ + h * DV_;
#pragma unroll
    for (int e = 0; e < 8; ++e) {
        int i = lane + e * 32;
        float gv = grow[i];
        outr[i] = f2bf(vals[e] * rs * gw[i] * (gv * sigmf(gv)));
    }
}

// =====================================================================
// 8) h = hidden + attn_out (fp32 out); y = rmsnorm(h)*w -> bf16
// =====================================================================
__global__ __launch_bounds__(256) void resid_rmsnorm_k(const float* __restrict__ hid,
                                                       const float* __restrict__ ao,
                                                       const float* __restrict__ w,
                                                       float* __restrict__ hout,
                                                       ushort_t* __restrict__ yb) {
    __shared__ float red[256];
    const int row = blockIdx.x;
    const int tid = threadIdx.x;
    const size_t base = (size_t)row * D_;
    float vals[D_ / 256];
    float s = 0.f;
#pragma unroll
    for (int e = 0; e < D_ / 256; ++e) {
        int c = tid + e * 256;
        float v = hid[base + c] + ao[base + c];
        vals[e] = v;
        hout[base + c] = v;
        s += v * v;
    }
    red[tid] = s; __syncthreads();
    for (int st = 128; st > 0; st >>= 1) {
        if (tid < st) red[tid] += red[tid + st];
        __syncthreads();
    }
    const float rs = rsqrtf(red[0] / (float)D_ + EPS_);
#pragma unroll
    for (int e = 0; e < D_ / 256; ++e) {
        int c = tid + e * 256;
        yb[base + c] = f2bf(vals[e] * rs * w[c]);
    }
}

// =====================================================================
// 9) swiglu: mid = silu(gy[:, :I]) * gy[:, I:] -> bf16
// =====================================================================
__global__ __launch_bounds__(256) void swiglu_k(const float* __restrict__ gy,
                                                ushort_t* __restrict__ mid) {
    size_t idx = (size_t)blockIdx.x * blockDim.x + threadIdx.x;
    if (idx >= (size_t)N_ * INTER_) return;
    size_t n = idx / INTER_;
    size_t j = idx % INTER_;
    const float* r = gy + n * (size_t)(2 * INTER_);
    mid[idx] = f2bf(siluf(r[j]) * r[INTER_ + j]);
}

// =====================================================================
// 10) out = h + down
// =====================================================================
__global__ __launch_bounds__(256) void add_k(const float* __restrict__ a,
                                             const float* __restrict__ b,
                                             float* __restrict__ out, size_t n) {
    size_t i = (size_t)blockIdx.x * blockDim.x + threadIdx.x;
    if (i < n) out[i] = a[i] + b[i];
}

// =====================================================================
// host-side orchestration
// =====================================================================
static inline void launch_gemm(const ushort_t* A, const ushort_t* Wt, float* C,
                               int M, int N, int K, hipStream_t stream) {
    dim3 grid(N / 128, M / 128);
    gemm_bf16_k<<<grid, 256, 0, stream>>>(A, Wt, C, M, N, K);
}

extern "C" void kernel_launch(void* const* d_in, const int* in_sizes, int n_in,
                              void* d_out, int out_size, void* d_ws, size_t ws_size,
                              hipStream_t stream) {
    const float* hidden  = (const float*)d_in[0];
    const float* anorm_w = (const float*)d_in[1];
    const float* Wq      = (const float*)d_in[2];
    const float* Wk      = (const float*)d_in[3];
    const float* Wv      = (const float*)d_in[4];
    const float* Wg      = (const float*)d_in[5];
    const float* Wo      = (const float*)d_in[6];
    const float* cwq     = (const float*)d_in[7];
    const float* cwk     = (const float*)d_in[8];
    const float* cwv     = (const float*)d_in[9];
    const float* gnorm_w = (const float*)d_in[10];
    const float* mnorm_w = (const float*)d_in[11];
    const float* Wgate   = (const float*)d_in[12];
    const float* Wdown   = (const float*)d_in[13];
    float* out = (float*)d_out;

    char* wsp = (char*)d_ws;
    size_t off = 0;
    auto take = [&](size_t bytes) -> char* {
        char* p = wsp + off;
        off += (bytes + 255) & ~(size_t)255;
        return p;
    };

    const size_t ND_f32 = (size_t)N_ * D_ * 4;
    const size_t ND_bf  = (size_t)N_ * D_ * 2;

    ushort_t* xb     = (ushort_t*)take(ND_bf);
    ushort_t* wqb    = (ushort_t*)take((size_t)D_ * D_ * 2);     // transposed bf16
    ushort_t* wkb    = (ushort_t*)take((size_t)D_ * D_ * 2);
    ushort_t* wvb    = (ushort_t*)take((size_t)D_ * D_ * 2);
    ushort_t* wgb    = (ushort_t*)take((size_t)D_ * D_ * 2);
    ushort_t* wob    = (ushort_t*)take((size_t)D_ * D_ * 2);
    ushort_t* wgateb = (ushort_t*)take((size_t)D_ * 2 * INTER_ * 2);
    ushort_t* wdownb = (ushort_t*)take((size_t)INTER_ * D_ * 2);
    float* qpre = (float*)take(ND_f32);
    float* kpre = (float*)take(ND_f32);
    float* vpre = (float*)take(ND_f32);
    float* qbuf = (float*)take(ND_f32);   // qbuf/kbuf/vbuf contiguous: reused as gy
    float* kbuf = (float*)take(ND_f32);
    float* vbuf = (float*)take(ND_f32);
    float* gbuf = (float*)take(ND_f32);
    float* obuf = (float*)take(ND_f32);
    ushort_t* ogb = (ushort_t*)take(ND_bf);
    (void)kbuf; (void)vbuf;

    // aliased reuse (all producers complete before reuse on the same stream)
    float*    attnout = qpre;              // dead after conv
    float*    hbuf    = kpre;              // dead after conv
    float*    down    = vpre;              // dead after conv
    float*    gy      = qbuf;              // spans q/k/v bufs (100MB >= 92MB)
    ushort_t* midb    = (ushort_t*)gbuf;   // dead after gated_out
    ushort_t* yb      = ogb;               // dead after Wo gemm

    // 1) x = rmsnorm(hidden) -> bf16
    rmsnorm_to_bf16_k<<<N_, 256, 0, stream>>>(hidden, anorm_w, xb);

    // 2) weights -> transposed bf16 (Wt[n][k]) for the TDM-fed GEMM
    auto conv_wT = [&](const float* s, ushort_t* d, int R, int Ccols) {
        dim3 grid(Ccols / 32, R / 32);
        transpose_to_bf16_k<<<grid, 256, 0, stream>>>(s, d, R, Ccols);
    };
    conv_wT(Wq, wqb, D_, D_);
    conv_wT(Wk, wkb, D_, D_);
    conv_wT(Wv, wvb, D_, D_);
    conv_wT(Wg, wgb, D_, D_);
    conv_wT(Wo, wob, D_, D_);
    conv_wT(Wgate, wgateb, D_, 2 * INTER_);
    conv_wT(Wdown, wdownb, INTER_, D_);

    // 3) projections
    launch_gemm(xb, wqb, qpre, N_, D_, D_, stream);
    launch_gemm(xb, wkb, kpre, N_, D_, D_, stream);
    launch_gemm(xb, wvb, vpre, N_, D_, D_, stream);
    launch_gemm(xb, wgb, gbuf, N_, D_, D_, stream);

    // 4) short conv + silu
    {
        int blocks = (int)(((size_t)N_ * D_ + 255) / 256);
        conv_silu_k<<<blocks, 256, 0, stream>>>(qpre, cwq, qbuf);
        conv_silu_k<<<blocks, 256, 0, stream>>>(kpre, cwk, kbuf);
        conv_silu_k<<<blocks, 256, 0, stream>>>(vpre, cwv, vbuf);
    }

    // 5) rope on q and k
    {
        int blocks = (int)(((size_t)B_ * T_ * H_ * (DK_ / 2) + 255) / 256);
        rope_k<<<blocks, 256, 0, stream>>>(qbuf);
        rope_k<<<blocks, 256, 0, stream>>>(kbuf);
    }

    // 6) chunked retention
    retention_k<<<B_ * H_ * (DV_ / 32), 256, 0, stream>>>(qbuf, kbuf, vbuf, obuf);

    // 7) group norm + gate -> bf16
    gated_out_k<<<N_, 256, 0, stream>>>(obuf, gbuf, gnorm_w, ogb);

    // 8) o @ Wo
    launch_gemm(ogb, wob, attnout, N_, D_, D_, stream);

    // 9) h = hidden + attn; y = rmsnorm(h) -> bf16
    resid_rmsnorm_k<<<N_, 256, 0, stream>>>(hidden, attnout, mnorm_w, hbuf, yb);

    // 10) gy = y @ W_gate
    launch_gemm(yb, wgateb, gy, N_, 2 * INTER_, D_, stream);

    // 11) swiglu -> bf16
    {
        int blocks = (int)(((size_t)N_ * INTER_ + 255) / 256);
        swiglu_k<<<blocks, 256, 0, stream>>>(gy, midb);
    }

    // 12) down = mid @ W_down
    launch_gemm(midb, wdownb, down, N_, D_, INTER_, stream);

    // 13) out = h + down
    {
        size_t n = (size_t)N_ * D_;
        int blocks = (int)((n + 255) / 256);
        add_k<<<blocks, 256, 0, stream>>>(hbuf, down, out, n);
    }
}
